// SquashedGaussianOnlySetActor_40003325395321
// MI455X (gfx1250) — matrix operand order repaired and checked by reading, not compiled, not run on hardware
//
#include <hip/hip_runtime.h>
#include <hip/hip_bf16.h>

typedef __attribute__((ext_vector_type(16))) _Float16     v16h;
typedef __attribute__((ext_vector_type(8)))  float        v8f;
typedef __attribute__((ext_vector_type(4)))  unsigned int v4u;

#define BATCH  512
#define NOBST  256
#define HID    256
#define XS     40    // x-tile row stride in halves (80B, 16B aligned)
#define HS     264   // h1-tile row stride in halves (528B, 16B aligned)

union FragU { v4u q[2]; v16h v; };

// A-fragment for V_WMMA_F32_16X16X32_F16 (16x32 f16, contiguous k = 16*sel + i, this code's order, same as B prep):
// lane row = lane&15; sel = lane>=16.
// V0-3: K (16*sel + 0..7) -> dwords [8*sel .. 8*sel+3]
// V4-7: K (16*sel + 8..15) -> dwords [8*sel+4 .. 8*sel+7]
__device__ __forceinline__ v16h load_a_frag(const _Float16* rowp, int sel) {
    const v4u* p = (const v4u*)rowp;   // rowp must be 16B aligned
    FragU u;
    u.q[0] = p[2 * sel];
    u.q[1] = p[2 * sel + 1];
    return u.v;
}

// B-fragment: pre-swizzled in memory as 8 consecutive dwords per lane.
__device__ __forceinline__ v16h load_b_frag(const unsigned int* dptr) {
    const v4u* p = (const v4u*)dptr;
    FragU u;
    u.q[0] = p[0];
    u.q[1] = p[1];
    return u.v;
}

__device__ __forceinline__ unsigned int pack2h(float a, float b) {
    union { _Float16 h[2]; unsigned int u; } p;
    p.h[0] = (_Float16)a; p.h[1] = (_Float16)b;
    return p.u;
}

// Low 32 bits of a generic pointer to a __shared__ object == LDS byte address
// (matches LLVM's flat->local addrspacecast lowering on amdgcn).
__device__ __forceinline__ unsigned lds_addr32(const void* p) {
    return (unsigned)(unsigned long long)p;
}

// ---------------------------------------------------------------------------
// Prep: convert W1 (19x256, K padded to 32) and W2 (256x256) to f16 in exact
// WMMA B-fragment order: frag dword index = ((ks*16 + nt)*32 + lane)*8 + j,
// element K = ks*32 + (lane<16 ? 2j : 16+2j), N = nt*16 + (lane&15).
// ---------------------------------------------------------------------------
__global__ __launch_bounds__(256) void prep_weights(
    const float* __restrict__ W1, const float* __restrict__ W2,
    unsigned int* __restrict__ w1h, unsigned int* __restrict__ w2h)
{
    const int idx = blockIdx.x * 256 + threadIdx.x;
    if (idx < 4096) {                     // W1: 16 nt-tiles, 1 k-step (K=32, rows >=19 zero)
        const int j    = idx & 7;
        const int lane = (idx >> 3) & 31;
        const int nt   = idx >> 8;
        const int s    = lane >> 4;
        const int k    = s * 16 + 2 * j;
        const int n    = nt * 16 + (lane & 15);
        const float lo = (k     < 19) ? W1[k * HID + n]       : 0.f;
        const float hi = (k + 1 < 19) ? W1[(k + 1) * HID + n] : 0.f;
        w1h[idx] = pack2h(lo, hi);
    }
    if (idx < 32768) {                    // W2: 8 k-steps x 16 nt-tiles
        const int j    = idx & 7;
        const int lane = (idx >> 3) & 31;
        const int nt   = (idx >> 8) & 15;
        const int ks   = idx >> 12;
        const int s    = lane >> 4;
        const int k    = ks * 32 + s * 16 + 2 * j;
        const int n    = nt * 16 + (lane & 15);
        w2h[idx] = pack2h(W2[k * HID + n], W2[(k + 1) * HID + n]);
    }
}

// ---------------------------------------------------------------------------
// Fused actor: one workgroup per batch element. 8 waves.
// ---------------------------------------------------------------------------
__global__ __launch_bounds__(256) void actor_fused(
    const float* __restrict__ obsg,  const float* __restrict__ obstg,
    const float* __restrict__ epsg,
    const float* __restrict__ b1g,   const float* __restrict__ b2g,
    const float* __restrict__ muWg,  const float* __restrict__ mubg,
    const float* __restrict__ lsWg,  const float* __restrict__ lsbg,
    const unsigned int* __restrict__ w1h, const unsigned int* __restrict__ w2h,
    float* __restrict__ outp)
{
    __shared__ __align__(16) unsigned int w2_lds[32768];   // 128 KB f16 B-frags
    __shared__ __align__(16) unsigned int w1_lds[4096];    // 16 KB
    __shared__ __align__(16) _Float16 x_lds[16 * XS];      // 16x32 x-tile (padded)
    __shared__ __align__(16) _Float16 h1_lds[16 * HS];     // 16x256 h1-tile (padded)
    __shared__ float mask_lds[NOBST];
    __shared__ float obs_lds[16];
    __shared__ float pooled_lds[HID];
    __shared__ float red_lds[4];

    const int tid  = threadIdx.x;
    const int w    = tid >> 5;       // wave 0..7
    const int lane = tid & 31;
    const int sel  = lane >> 4;      // 0: lanes 0-15, 1: lanes 16-31
    const int b    = blockIdx.x;

    // ---- stage weights into LDS with async VMEM->LDS copies (no VGPR round
    // trip, ASYNCcnt-tracked; overlaps the mask/obs loads below) ----
    {
        const unsigned lds2 = lds_addr32(w2_lds);
        #pragma unroll 4
        for (int i = tid; i < 8192; i += 256) {
            const unsigned voff = (unsigned)i * 16u;
            asm volatile("global_load_async_to_lds_b128 %0, %1, %2 offset:0"
                         :: "v"(lds2 + voff), "v"(voff), "s"(w2h)
                         : "memory");
        }
        const unsigned lds1 = lds_addr32(w1_lds);
        #pragma unroll 2
        for (int i = tid; i < 1024; i += 256) {
            const unsigned voff = (unsigned)i * 16u;
            asm volatile("global_load_async_to_lds_b128 %0, %1, %2 offset:0"
                         :: "v"(lds1 + voff), "v"(voff), "s"(w1h)
                         : "memory");
        }
    }
    mask_lds[tid] = obstg[b * 1280 + 4 * 256 + tid];
    if (tid < 16) obs_lds[tid] = (tid < 15) ? obsg[b * 15 + tid] : 0.f;
    asm volatile("s_wait_asynccnt 0x0" ::: "memory");
    __syncthreads();

    const int colA = w * 16 + (lane & 15);
    const int colB = (w + 8) * 16 + (lane & 15);
    const float b1A = b1g[colA], b1B = b1g[colB];
    const float b2A = b2g[colA], b2B = b2g[colB];
    float pp0 = 0.f, pp1 = 0.f;   // mask-weighted pooled partials for colA/colB

    for (int mt = 0; mt < 16; ++mt) {
        // ---- build x-tile: rows = obstacles 16*mt..16*mt+15, cols 0..31 f16 ----
        for (int idx = tid; idx < 512; idx += 256) {
            const int r = idx >> 5, c = idx & 31;
            float v = 0.f;
            if (c < 15)      v = obs_lds[c];
            else if (c < 19) v = obstg[b * 1280 + (c - 15) * 256 + 16 * mt + r];
            x_lds[r * XS + c] = (_Float16)v;
        }
        __syncthreads();

        // ---- layer 1: 16x32 @ 32x256, each wave does 2 N-tiles (shared A) ----
        const v16h xa = load_a_frag(x_lds + (lane & 15) * XS, sel);
        #pragma unroll
        for (int t = 0; t < 2; ++t) {
            const int nt = w + 8 * t;
            const v16h bf = load_b_frag(w1_lds + (nt * 32 + lane) * 8);
            v8f c = {};
            c = __builtin_amdgcn_wmma_f32_16x16x32_f16(false, xa, false, bf,
                                                       (short)0, c, false, false);
            const int   col = (t == 0) ? colA : colB;
            const float bv  = (t == 0) ? b1A  : b1B;
            #pragma unroll
            for (int r = 0; r < 8; ++r) {
                float v = c[r] + bv;
                v = v > 0.f ? v : 0.f;
                h1_lds[(8 * sel + r) * HS + col] = (_Float16)v;
            }
        }
        __syncthreads();

        // ---- layer 2: 16x256 @ 256x256, ks outer so each A-frag is loaded
        // once and feeds two back-to-back WMMAs (one per N-tile) ----
        {
            v8f acc0 = {b2A, b2A, b2A, b2A, b2A, b2A, b2A, b2A};
            v8f acc1 = {b2B, b2B, b2B, b2B, b2B, b2B, b2B, b2B};
            #pragma unroll
            for (int ks = 0; ks < 8; ++ks) {
                const v16h a   = load_a_frag(h1_lds + (lane & 15) * HS + 32 * ks, sel);
                const v16h bf0 = load_b_frag(w2_lds + ((ks * 16 + w) * 32 + lane) * 8);
                acc0 = __builtin_amdgcn_wmma_f32_16x16x32_f16(false, a, false, bf0,
                                                              (short)0, acc0, false, false);
                const v16h bf1 = load_b_frag(w2_lds + ((ks * 16 + w + 8) * 32 + lane) * 8);
                acc1 = __builtin_amdgcn_wmma_f32_16x16x32_f16(false, a, false, bf1,
                                                              (short)0, acc1, false, false);
            }
            // C layout: row = 8*sel + r, col fixed per lane -> pool in registers
            float s0 = 0.f, s1 = 0.f;
            #pragma unroll
            for (int r = 0; r < 8; ++r) {
                const float m  = mask_lds[16 * mt + 8 * sel + r];
                const float h0 = acc0[r] > 0.f ? acc0[r] : 0.f;
                const float h1 = acc1[r] > 0.f ? acc1[r] : 0.f;
                s0 += m * h0;
                s1 += m * h1;
            }
            pp0 += s0;
            pp1 += s1;
        }
        __syncthreads();   // protect x_lds/h1_lds reuse next iteration
    }

    // merge the two half-tile partials (lanes l and l^16 own the same column)
    pp0 += __shfl_xor(pp0, 16, 32);
    pp1 += __shfl_xor(pp1, 16, 32);
    if (lane < 16) {
        pooled_lds[colA] = pp0;
        pooled_lds[colB] = pp1;
    }
    __syncthreads();

    // ---- head: waves 0..3 compute {mu0, mu1, ls0, ls1} = pooled . W ----
    if (w < 4) {
        const float* Wsel = (w < 2) ? muWg : lsWg;
        const int j = w & 1;
        float s = 0.f;
        #pragma unroll
        for (int i = 0; i < 8; ++i) {
            const int hh = lane + 32 * i;
            s += pooled_lds[hh] * Wsel[hh * 2 + j];
        }
        #pragma unroll
        for (int off = 16; off >= 1; off >>= 1) s += __shfl_xor(s, off, 32);
        if (lane == 0) red_lds[w] = s;
    }
    __syncthreads();

    if (tid == 0) {
        const float mu0 = red_lds[0] + mubg[0];
        const float mu1 = red_lds[1] + mubg[1];
        float l0 = fminf(fmaxf(red_lds[2] + lsbg[0], -20.f), 2.f);
        float l1 = fminf(fmaxf(red_lds[3] + lsbg[1], -20.f), 2.f);
        const float e0 = epsg[b * 2 + 0], e1 = epsg[b * 2 + 1];
        const float a0 = mu0 + __expf(l0) * e0;
        const float a1 = mu1 + __expf(l1) * e1;
        float logp = -0.5f * (e0 * e0 + e1 * e1) - (l0 + l1)
                     - 1.8378770664093453f;                 // log(2*pi)
        const float LN2 = 0.6931471805599453f;
        auto sp = [](float x) {                              // stable softplus
            return fmaxf(x, 0.f) + log1pf(__expf(-fabsf(x)));
        };
        logp -= 2.f * (LN2 - a0 - sp(-2.f * a0));
        logp -= 2.f * (LN2 - a1 - sp(-2.f * a1));
        outp[b * 2 + 0]     = tanhf(a0);
        outp[b * 2 + 1]     = tanhf(a1);
        outp[2 * BATCH + b] = logp;                          // logp block after pi
    }
}

extern "C" void kernel_launch(void* const* d_in, const int* in_sizes, int n_in,
                              void* d_out, int out_size, void* d_ws, size_t ws_size,
                              hipStream_t stream) {
    const float* obs  = (const float*)d_in[0];
    const float* obst = (const float*)d_in[1];
    const float* eps  = (const float*)d_in[2];
    const float* W1   = (const float*)d_in[3];
    const float* b1   = (const float*)d_in[4];
    const float* W2   = (const float*)d_in[5];
    const float* b2   = (const float*)d_in[6];
    const float* muW  = (const float*)d_in[7];
    const float* mub  = (const float*)d_in[8];
    const float* lsW  = (const float*)d_in[9];
    const float* lsb  = (const float*)d_in[10];

    unsigned int* w1h = (unsigned int*)d_ws;       // 4096 dwords
    unsigned int* w2h = w1h + 4096;                // 32768 dwords (total 144 KB)

    prep_weights<<<128, 256, 0, stream>>>(W1, W2, w1h, w2h);
    actor_fused<<<BATCH, 256, 0, stream>>>(obs, obst, eps, b1, b2,
                                           muW, mub, lsW, lsb,
                                           w1h, w2h, (float*)d_out);
}
